// Seq2seq_no_attention_56444460204275
// MI455X (gfx1250) — matrix operand order repaired and checked by reading, not compiled, not run on hardware
//
#include <hip/hip_runtime.h>

typedef _Float16 f16;
typedef __attribute__((ext_vector_type(16))) _Float16 v16h;
typedef __attribute__((ext_vector_type(8)))  _Float16 v8h;
typedef __attribute__((ext_vector_type(8)))  float    v8f;

#define WMMA16(a, b, c) \
  __builtin_amdgcn_wmma_f32_16x16x32_f16(false, (a), false, (b), (short)0, (c), false, false)

namespace {
constexpr int Bv = 32, Sv = 64, Tv = 64, Hv = 512, Ev = 512, Fv = 2048, Vv = 32000;
constexpr int G3H = 3 * Hv;   // 1536
constexpr int MP  = 2048;     // padded row count for decoder-side (2016-row) matrices
}

// ---------------- fragment loaders (per ISA 7.12.2 VGPR layouts) ----------------
// A (16x32 f16): lane L -> row = L&15; VGPR0-3: K = kb..kb+7, VGPR4-7: K = kb+16..kb+23, kb = (L>=16)?8:0
// B (32x16 f16): lane L -> col = L&15; VGPR0-3: K = kb..kb+7, VGPR4-7: K = kb+16..kb+23, kb = (L>=16)?8:0
__device__ __forceinline__ v16h ld_pair(const f16* p0, const f16* p1) {
  union { v16h v; v8h h[2]; } u;
  u.h[0] = *(const v8h*)p0;
  u.h[1] = *(const v8h*)p1;
  return u.v;
}
__device__ __forceinline__ v16h ld_fragA(const f16* A, int lda, int row, int k0, int lane) {
  int kb = (lane >> 4) << 3;
  const f16* base = A + (size_t)row * lda + k0 + kb;
  return ld_pair(base, base + 16);
}
__device__ __forceinline__ v16h ld_fragB(const f16* W, int ldb, int n, int k0, int lane) {
  int kb = (lane >> 4) << 3;
  const f16* base = W + (size_t)n * ldb + k0 + kb;
  return ld_pair(base, base + 16);
}

// ---------------- generic GEMM: C[M,N] = A[M,K] @ W[N,K]^T + bias ----------------
// 2x2 register tiling per wave: wave computes 32x32; WG (8 waves, 2m x 4n) = 64x128.
// grid.x = Npad/128, grid.y = Mpad/64. Stores guarded by m < M; A must be padded to Mpad rows.
__global__ __launch_bounds__(256) void gemm_xwt(
    const f16* __restrict__ A, int lda, const f16* __restrict__ W, int K,
    const float* __restrict__ bias, float* __restrict__ Cf, f16* __restrict__ Ch,
    int ldc, int relu, int M) {
  int lane = threadIdx.x & 31;
  int w = threadIdx.x >> 5;
  int wn = w & 3, wm = w >> 2;
  int m0 = blockIdx.y * 64 + wm * 32;
  int n0 = blockIdx.x * 128 + wn * 32;
  int r15 = lane & 15;
  v8f acc00 = {}, acc01 = {}, acc10 = {}, acc11 = {};
  for (int k0 = 0; k0 < K; k0 += 32) {
    v16h a0 = ld_fragA(A, lda, m0 + r15, k0, lane);
    v16h a1 = ld_fragA(A, lda, m0 + 16 + r15, k0, lane);
    v16h b0 = ld_fragB(W, K, n0 + r15, k0, lane);
    v16h b1 = ld_fragB(W, K, n0 + 16 + r15, k0, lane);
    acc00 = WMMA16(a0, b0, acc00);
    acc01 = WMMA16(a0, b1, acc01);
    acc10 = WMMA16(a1, b0, acc10);
    acc11 = WMMA16(a1, b1, acc11);
  }
  int hi8 = (lane >> 4) << 3;
#pragma unroll
  for (int mt = 0; mt < 2; ++mt) {
#pragma unroll
    for (int nt = 0; nt < 2; ++nt) {
      v8f acc = (mt == 0) ? (nt == 0 ? acc00 : acc01) : (nt == 0 ? acc10 : acc11);
      int n = n0 + nt * 16 + r15;
      float bv = bias ? bias[n] : 0.f;
#pragma unroll
      for (int r = 0; r < 8; ++r) {
        int m = m0 + mt * 16 + hi8 + r;
        if (m < M) {
          float v = acc[r] + bv;
          if (relu) v = fmaxf(v, 0.f);
          if (Cf) Cf[(size_t)m * ldc + n] = v;
          if (Ch) Ch[(size_t)m * ldc + n] = (f16)v;
        }
      }
    }
  }
}

// ---------------- classifier: logits = out[2016,512] @ embW[32000,512]^T + cls_b ----------------
// Same 2x2 tiling; row r = t*32+b -> d_out[b][t+1][n].
__global__ __launch_bounds__(256) void classifier_gemm(
    const f16* __restrict__ A, const f16* __restrict__ W,
    const float* __restrict__ clsb, float* __restrict__ out) {
  int lane = threadIdx.x & 31;
  int w = threadIdx.x >> 5;
  int wn = w & 3, wm = w >> 2;
  int m0 = blockIdx.y * 64 + wm * 32;
  int n0 = blockIdx.x * 128 + wn * 32;
  int r15 = lane & 15;
  v8f acc00 = {}, acc01 = {}, acc10 = {}, acc11 = {};
  for (int k0 = 0; k0 < Hv; k0 += 32) {
    v16h a0 = ld_fragA(A, Hv, m0 + r15, k0, lane);
    v16h a1 = ld_fragA(A, Hv, m0 + 16 + r15, k0, lane);
    v16h b0 = ld_fragB(W, Hv, n0 + r15, k0, lane);
    v16h b1 = ld_fragB(W, Hv, n0 + 16 + r15, k0, lane);
    acc00 = WMMA16(a0, b0, acc00);
    acc01 = WMMA16(a0, b1, acc01);
    acc10 = WMMA16(a1, b0, acc10);
    acc11 = WMMA16(a1, b1, acc11);
  }
  int hi8 = (lane >> 4) << 3;
  const int Mv = (Tv - 1) * Bv;  // 2016
#pragma unroll
  for (int mt = 0; mt < 2; ++mt) {
#pragma unroll
    for (int nt = 0; nt < 2; ++nt) {
      v8f acc = (mt == 0) ? (nt == 0 ? acc00 : acc01) : (nt == 0 ? acc10 : acc11);
      int n = n0 + nt * 16 + r15;
      float bv = clsb[n];
#pragma unroll
      for (int r = 0; r < 8; ++r) {
        int m = m0 + mt * 16 + hi8 + r;
        if (m < Mv) {
          int b = m & 31, t = m >> 5;
          out[((size_t)(b * Tv + t + 1)) * Vv + n] = acc[r] + bv;
        }
      }
    }
  }
}

// ---------------- fused GRU step: gh = h@Whh^T+bhh, gate math, hnew ----------------
// hnew[32,512]; 64 wave-tiles (2 M x 32 N of 16); grid = 8 WGs x 256 thr.
__global__ __launch_bounds__(256) void gru_step(
    const f16* __restrict__ hA, const f16* __restrict__ Whh,
    const float* __restrict__ bhh, const float* __restrict__ xp,
    const float* __restrict__ hin, float* __restrict__ hout,
    f16* __restrict__ hAout, f16* __restrict__ yout, int y_ld, int y_off) {
  int lane = threadIdx.x & 31;
  int w = threadIdx.x >> 5;
  int g = blockIdx.x * 8 + w;
  int mtile = g & 1;
  int ntile = g >> 1;
  int rowA = mtile * 16 + (lane & 15);
  int n = ntile * 16 + (lane & 15);
  v8f accR = {}, accZ = {}, accN = {};
  for (int k0 = 0; k0 < Hv; k0 += 32) {
    v16h a = ld_fragA(hA, Hv, rowA, k0, lane);
    v16h br = ld_fragB(Whh, Hv, n, k0, lane);
    v16h bz = ld_fragB(Whh, Hv, n + Hv, k0, lane);
    v16h bn = ld_fragB(Whh, Hv, n + 2 * Hv, k0, lane);
    accR = WMMA16(a, br, accR);
    accZ = WMMA16(a, bz, accZ);
    accN = WMMA16(a, bn, accN);
  }
  float bhr = bhh[n], bhz = bhh[n + Hv], bhn = bhh[n + 2 * Hv];
  int mBase = mtile * 16 + ((lane >> 4) << 3);
#pragma unroll
  for (int r = 0; r < 8; ++r) {
    int b = mBase + r;
    float xr = xp[b * G3H + n];
    float xz = xp[b * G3H + n + Hv];
    float xn = xp[b * G3H + n + 2 * Hv];
    float rg = 1.f / (1.f + __expf(-(xr + accR[r] + bhr)));
    float zg = 1.f / (1.f + __expf(-(xz + accZ[r] + bhz)));
    float ng = tanhf(xn + rg * (accN[r] + bhn));
    float hv = (1.f - zg) * ng + zg * hin[b * Hv + n];
    hout[b * Hv + n] = hv;
    f16 hh = (f16)hv;
    hAout[b * Hv + n] = hh;
    yout[(size_t)b * y_ld + y_off + n] = hh;
  }
}

// ---------------- small utility kernels ----------------
__global__ void cast_f32_f16(const float* __restrict__ s, f16* __restrict__ d, int n) {
  int i = blockIdx.x * blockDim.x + threadIdx.x;
  if (i < n) d[i] = (f16)s[i];
}
__global__ void zero_f32_k(float* p, int n) {
  int i = blockIdx.x * blockDim.x + threadIdx.x;
  if (i < n) p[i] = 0.f;
}
__global__ void zero_f16_k(f16* p, int n) {
  int i = blockIdx.x * blockDim.x + threadIdx.x;
  if (i < n) p[i] = (f16)0.f;
}
// dst[(t*32+b)*512 + e] = f16(embW[tok[b*64+t]*512 + e])
__global__ void embed_gather(const int* __restrict__ tok, const float* __restrict__ embW,
                             f16* __restrict__ dst, int steps) {
  int i = blockIdx.x * blockDim.x + threadIdx.x;
  if (i >= steps * Bv * Ev) return;
  int e = i & (Ev - 1);
  int row = i >> 9;
  int b = row & 31, t = row >> 5;
  dst[i] = (f16)embW[(size_t)tok[b * Tv + t] * Ev + e];
}
__global__ void copy_to_last(const f16* __restrict__ h, f16* __restrict__ last, int off) {
  int i = blockIdx.x * blockDim.x + threadIdx.x;
  if (i >= Bv * Hv) return;
  int k = i & (Hv - 1), b = i >> 9;
  last[b * 2 * Hv + off + k] = h[i];
}
__global__ void zero_out_slice(float* __restrict__ out) {
  int i = blockIdx.x * blockDim.x + threadIdx.x;
  if (i >= Bv * Vv) return;
  int b = i / Vv, v = i - b * Vv;
  out[(size_t)b * Tv * Vv + v] = 0.f;
}

// ---------------- host orchestration ----------------
extern "C" void kernel_launch(void* const* d_in, const int* in_sizes, int n_in,
                              void* d_out, int out_size, void* d_ws, size_t ws_size,
                              hipStream_t stream) {
  (void)in_sizes; (void)n_in; (void)out_size; (void)ws_size;
  // params flattened in insertion order
  const int* src_tok = (const int*)d_in[0];
  const int* tgt_tok = (const int*)d_in[1];
  const float* embW  = (const float*)d_in[3];
  const float* cls_b = (const float*)d_in[4];
  struct GRUW { const float *Wih, *Whh, *bih, *bhh; };
  GRUW encw[2][2], decw[2];
  int idx = 5;
  for (int l = 0; l < 2; ++l)
    for (int d = 0; d < 2; ++d) {
      encw[l][d].Wih = (const float*)d_in[idx++];
      encw[l][d].Whh = (const float*)d_in[idx++];
      encw[l][d].bih = (const float*)d_in[idx++];
      encw[l][d].bhh = (const float*)d_in[idx++];
    }
  for (int l = 0; l < 2; ++l) {
    decw[l].Wih = (const float*)d_in[idx++];
    decw[l].Whh = (const float*)d_in[idx++];
    decw[l].bih = (const float*)d_in[idx++];
    decw[l].bhh = (const float*)d_in[idx++];
  }
  const float* ffW1 = (const float*)d_in[idx++];
  const float* ffb1 = (const float*)d_in[idx++];
  const float* ffW2 = (const float*)d_in[idx++];
  const float* ffb2 = (const float*)d_in[idx++];
  const float* ffwW = (const float*)d_in[idx++];
  const float* ffwb = (const float*)d_in[idx++];

  // workspace bump allocator (256B aligned)
  char* wp = (char*)d_ws;
  auto alloc = [&](size_t bytes) -> void* {
    void* r = (void*)wp;
    wp += (bytes + 255) & ~(size_t)255;
    return r;
  };
  f16* emb_h = (f16*)alloc((size_t)Vv * Ev * 2);
  f16 *encWih_h[2][2], *encWhh_h[2][2];
  for (int l = 0; l < 2; ++l)
    for (int d = 0; d < 2; ++d) {
      int din = (l == 0) ? Ev : 2 * Hv;
      encWih_h[l][d] = (f16*)alloc((size_t)G3H * din * 2);
      encWhh_h[l][d] = (f16*)alloc((size_t)G3H * Hv * 2);
    }
  f16 *decWih_h[2], *decWhh_h[2];
  for (int l = 0; l < 2; ++l) {
    decWih_h[l] = (f16*)alloc((size_t)G3H * Hv * 2);
    decWhh_h[l] = (f16*)alloc((size_t)G3H * Hv * 2);
  }
  f16* ffW1_h = (f16*)alloc((size_t)Fv * 2 * Hv * 2);
  f16* ffW2_h = (f16*)alloc((size_t)Hv * Fv * 2);
  f16* ffwW_h = (f16*)alloc((size_t)Hv * Hv * 2);
  f16* src_emb = (f16*)alloc((size_t)Sv * Bv * Ev * 2);
  f16* xs0 = (f16*)alloc((size_t)Sv * Bv * 2 * Hv * 2);
  f16* xs1 = (f16*)alloc((size_t)Sv * Bv * 2 * Hv * 2);
  float* xpf = (float*)alloc((size_t)Sv * Bv * G3H * 4);
  float* xpb = (float*)alloc((size_t)Sv * Bv * G3H * 4);
  float* h32A = (float*)alloc((size_t)Bv * Hv * 4);
  float* h32B = (float*)alloc((size_t)Bv * Hv * 4);
  f16* h16A = (f16*)alloc((size_t)Bv * Hv * 2);
  f16* h16B = (f16*)alloc((size_t)Bv * Hv * 2);
  f16* last_h = (f16*)alloc((size_t)64 * 2 * Hv * 2);       // padded to 64 rows
  f16* ff1_h  = (f16*)alloc((size_t)64 * Fv * 2);           // padded to 64 rows
  float* ctx_f = (float*)alloc((size_t)Bv * Hv * 4);
  f16* ctx_h = (f16*)alloc((size_t)Bv * Hv * 2);
  f16* dec_in = (f16*)alloc((size_t)MP * Ev * 2);           // padded to 2048 rows
  f16* dec_y0 = (f16*)alloc((size_t)MP * Hv * 2);
  f16* dec_y1 = (f16*)alloc((size_t)MP * Hv * 2);
  f16* out_h  = (f16*)alloc((size_t)MP * Hv * 2);

  auto cast = [&](const float* s, f16* d, size_t n) {
    cast_f32_f16<<<(unsigned)((n + 255) / 256), 256, 0, stream>>>(s, d, (int)n);
  };
  auto zpad = [&](f16* p, int n) {
    zero_f16_k<<<(n + 255) / 256, 256, 0, stream>>>(p, n);
  };
  // weight casts
  cast(embW, emb_h, (size_t)Vv * Ev);
  for (int l = 0; l < 2; ++l)
    for (int d = 0; d < 2; ++d) {
      int din = (l == 0) ? Ev : 2 * Hv;
      cast(encw[l][d].Wih, encWih_h[l][d], (size_t)G3H * din);
      cast(encw[l][d].Whh, encWhh_h[l][d], (size_t)G3H * Hv);
    }
  for (int l = 0; l < 2; ++l) {
    cast(decw[l].Wih, decWih_h[l], (size_t)G3H * Hv);
    cast(decw[l].Whh, decWhh_h[l], (size_t)G3H * Hv);
  }
  cast(ffW1, ffW1_h, (size_t)Fv * 2 * Hv);
  cast(ffW2, ffW2_h, (size_t)Hv * Fv);
  cast(ffwW, ffwW_h, (size_t)Hv * Hv);
  // zero pad rows so padded-tile reads are defined
  const int Td = Tv - 1;  // 63
  zpad(dec_in + (size_t)Td * Bv * Ev, (MP - Td * Bv) * Ev);
  zpad(dec_y0 + (size_t)Td * Bv * Hv, (MP - Td * Bv) * Hv);
  zpad(dec_y1 + (size_t)Td * Bv * Hv, (MP - Td * Bv) * Hv);
  zpad(out_h  + (size_t)Td * Bv * Hv, (MP - Td * Bv) * Hv);
  zpad(last_h + (size_t)Bv * 2 * Hv, 32 * 2 * Hv);
  zpad(ff1_h  + (size_t)Bv * Fv, 32 * Fv);

  // ---- encoder ----
  embed_gather<<<(Sv * Bv * Ev + 255) / 256, 256, 0, stream>>>(src_tok, embW, src_emb, Sv);
  const f16* lin = src_emb;
  int din = Ev;
  f16* louts[2] = {xs0, xs1};
  for (int l = 0; l < 2; ++l) {
    f16* lout = louts[l];
    dim3 gproj(G3H / 128, (Sv * Bv) / 64);
    gemm_xwt<<<gproj, 256, 0, stream>>>(lin, din, encWih_h[l][0], din, encw[l][0].bih,
                                        xpf, nullptr, G3H, 0, Sv * Bv);
    gemm_xwt<<<gproj, 256, 0, stream>>>(lin, din, encWih_h[l][1], din, encw[l][1].bih,
                                        xpb, nullptr, G3H, 0, Sv * Bv);
    for (int d = 0; d < 2; ++d) {
      float* xp = d ? xpb : xpf;
      zero_f32_k<<<(Bv * Hv + 255) / 256, 256, 0, stream>>>(h32A, Bv * Hv);
      zero_f16_k<<<(Bv * Hv + 255) / 256, 256, 0, stream>>>(h16A, Bv * Hv);
      float *hc = h32A, *hn = h32B;
      f16 *hc16 = h16A, *hn16 = h16B;
      for (int t = 0; t < Sv; ++t) {
        int ti = d ? (Sv - 1 - t) : t;
        gru_step<<<8, 256, 0, stream>>>(hc16, encWhh_h[l][d], encw[l][d].bhh,
                                        xp + (size_t)ti * Bv * G3H, hc, hn, hn16,
                                        lout + (size_t)ti * Bv * 2 * Hv, 2 * Hv, d * Hv);
        float* tf = hc; hc = hn; hn = tf;
        f16* th = hc16; hc16 = hn16; hn16 = th;
      }
      if (l == 1)
        copy_to_last<<<(Bv * Hv + 255) / 256, 256, 0, stream>>>(hc16, last_h, d * Hv);
    }
    lin = lout;
    din = 2 * Hv;
  }

  // ---- ff: ctx = relu(last @ W1^T + b1) @ W2^T + b2 ----
  gemm_xwt<<<dim3(Fv / 128, 1), 256, 0, stream>>>(last_h, 2 * Hv, ffW1_h, 2 * Hv, ffb1,
                                                  nullptr, ff1_h, Fv, 1, Bv);
  gemm_xwt<<<dim3(Hv / 128, 1), 256, 0, stream>>>(ff1_h, Fv, ffW2_h, Fv, ffb2,
                                                  ctx_f, ctx_h, Hv, 0, Bv);

  // ---- decoder ----
  embed_gather<<<(Td * Bv * Ev + 255) / 256, 256, 0, stream>>>(tgt_tok, embW, dec_in, Td);
  const f16* dlin = dec_in;
  f16* youts[2] = {dec_y0, dec_y1};
  for (int l = 0; l < 2; ++l) {
    dim3 gproj(G3H / 128, MP / 64);
    gemm_xwt<<<gproj, 256, 0, stream>>>(dlin, Hv, decWih_h[l], Hv, decw[l].bih,
                                        xpf, nullptr, G3H, 0, Td * Bv);
    hipMemcpyAsync(h32A, ctx_f, (size_t)Bv * Hv * 4, hipMemcpyDeviceToDevice, stream);
    hipMemcpyAsync(h16A, ctx_h, (size_t)Bv * Hv * 2, hipMemcpyDeviceToDevice, stream);
    float *hc = h32A, *hn = h32B;
    f16 *hc16 = h16A, *hn16 = h16B;
    for (int t = 0; t < Td; ++t) {
      gru_step<<<8, 256, 0, stream>>>(hc16, decWhh_h[l], decw[l].bhh,
                                      xpf + (size_t)t * Bv * G3H, hc, hn, hn16,
                                      youts[l] + (size_t)t * Bv * Hv, Hv, 0);
      float* tf = hc; hc = hn; hn = tf;
      f16* th = hc16; hc16 = hn16; hn16 = th;
    }
    dlin = youts[l];
  }

  // ---- out = y @ ffw_W^T + ffw_b (f16 for classifier A) ----
  gemm_xwt<<<dim3(Hv / 128, MP / 64), 256, 0, stream>>>(
      dec_y1, Hv, ffwW_h, Hv, ffwb, nullptr, out_h, Hv, 0, Td * Bv);

  // ---- logits = out @ embW^T + cls_b -> d_out[b][t+1][v]; slot t=0 zeroed ----
  float* dout = (float*)d_out;
  zero_out_slice<<<(Bv * Vv + 255) / 256, 256, 0, stream>>>(dout);
  classifier_gemm<<<dim3(Vv / 128, MP / 64), 256, 0, stream>>>(out_h, emb_h, cls_b, dout);
}
